// PointNetSAModule_71691594104932
// MI455X (gfx1250) — compile-verified, matrix operand order checked by reading, not run on hardware
//
#include <hip/hip_runtime.h>
#include <hip/hip_bf16.h>

// ---------------------------------------------------------------------------
// PointNet++ Set Abstraction for MI455X (gfx1250, wave32, WMMA)
// B=16, N=4096, Cpts=32, S=1024, ns=32, MLP 35->64->64->128, BN(train)+ReLU
// ---------------------------------------------------------------------------

#define B_      16
#define N_      4096
#define CPTS_   32
#define S_      1024
#define NS_     32
#define NGROUP  (B_ * S_)              // 16384 centroids
#define MROWS   (NGROUP * NS_)         // 524288 gemm rows
#define KPAD    64                     // K padded to 64 (2 wmma k-steps)
#define R2_     0.04f                  // RADIUS^2
#define BN_EPS_ 1e-3f

typedef __attribute__((ext_vector_type(16))) __bf16    v16bf;
typedef __attribute__((ext_vector_type(8)))  float     v8f;
typedef __attribute__((ext_vector_type(4)))  unsigned  uint4v;

struct bfpack { uint4v lo, hi; };      // 32 bytes == v16bf

__device__ __forceinline__ unsigned short f2bf_u(float f) {
    unsigned u = __builtin_bit_cast(unsigned, f);
    u = (u + 0x7FFFu + ((u >> 16) & 1u)) >> 16;   // round-to-nearest-even
    return (unsigned short)u;
}
__device__ __forceinline__ __bf16 u2bf(unsigned short u) {
    return __builtin_bit_cast(__bf16, u);
}

// ---------------------------------------------------------------------------
// 1) Farthest point sampling: one 1024-thread block per batch.
//    dists live in registers (4 pts/thread, contiguous ownership so the
//    reduction tie-break order matches jnp.argmax first-index semantics).
//    2 barriers per iteration.
// ---------------------------------------------------------------------------
__global__ void fps_kernel(const float* __restrict__ xyz,
                           float* __restrict__ new_xyz) {
    const int b    = blockIdx.x;
    const int t    = threadIdx.x;           // 0..1023
    const int lane = t & 31, wv = t >> 5;
    const float* X = xyz + (size_t)b * N_ * 3;

    float px[4], py[4], pz[4], dist[4];
#pragma unroll
    for (int i = 0; i < 4; ++i) {
        const int p = t * 4 + i;
        px[i] = X[p * 3 + 0];
        py[i] = X[p * 3 + 1];
        pz[i] = X[p * 3 + 2];
        dist[i] = 1e10f;
    }

    __shared__ float s_val[32];
    __shared__ int   s_idx[32];
    __shared__ float s_c[3];

    if (t == 0) {                            // first sample is index 0
        s_c[0] = X[0]; s_c[1] = X[1]; s_c[2] = X[2];
        float* o = new_xyz + (size_t)b * S_ * 3;
        o[0] = X[0]; o[1] = X[1]; o[2] = X[2];
    }
    __syncthreads();

    for (int s = 0; s < S_; ++s) {
        const float cx = s_c[0], cy = s_c[1], cz = s_c[2];

        float bv = -1.0f; int bi = 0;
#pragma unroll
        for (int i = 0; i < 4; ++i) {
            const float dx = px[i] - cx, dy = py[i] - cy, dz = pz[i] - cz;
            const float d  = dx * dx + dy * dy + dz * dz;
            dist[i] = fminf(dist[i], d);
            if (dist[i] > bv) { bv = dist[i]; bi = t * 4 + i; }
        }
        // wave32 argmax reduction (lower index wins ties)
#pragma unroll
        for (int off = 16; off > 0; off >>= 1) {
            const float ov = __shfl_down(bv, off);
            const int   oi = __shfl_down(bi, off);
            if (ov > bv || (ov == bv && oi < bi)) { bv = ov; bi = oi; }
        }
        if (lane == 0) { s_val[wv] = bv; s_idx[wv] = bi; }
        __syncthreads();
        if (wv == 0) {
            bv = s_val[lane]; bi = s_idx[lane];
#pragma unroll
            for (int off = 16; off > 0; off >>= 1) {
                const float ov = __shfl_down(bv, off);
                const int   oi = __shfl_down(bi, off);
                if (ov > bv || (ov == bv && oi < bi)) { bv = ov; bi = oi; }
            }
            if (lane == 0 && s + 1 < S_) {
                const float* c = X + (size_t)bi * 3;
                s_c[0] = c[0]; s_c[1] = c[1]; s_c[2] = c[2];
                float* o = new_xyz + ((size_t)b * S_ + s + 1) * 3;
                o[0] = c[0]; o[1] = c[1]; o[2] = c[2];
            }
        }
        __syncthreads();
    }
}

// ---------------------------------------------------------------------------
// 2) query_ball: block stages its batch's 48KB point cloud in LDS, then one
//    wave per centroid scans it; ballot + prefix popcount keeps the first-32
//    in-ball indices in ascending order (matches reference top_k ordering).
// ---------------------------------------------------------------------------
__global__ void ball_kernel(const float* __restrict__ xyz,
                            const float* __restrict__ new_xyz,
                            int* __restrict__ idx_ws,
                            int* __restrict__ out_idx) {
    __shared__ float sx[N_], sy[N_], sz[N_];   // 48 KB of 320 KB LDS
    __shared__ int   s_hits[8][NS_];

    const int wv   = threadIdx.x >> 5;
    const int lane = threadIdx.x & 31;
    const int gid  = blockIdx.x * 8 + wv;      // centroid id, < 16384
    const int b    = gid >> 10;                // block-uniform (8 | 1024)
    const float* X = xyz + (size_t)b * N_ * 3;

    for (int i = threadIdx.x; i < N_; i += 256) {
        sx[i] = X[i * 3 + 0];
        sy[i] = X[i * 3 + 1];
        sz[i] = X[i * 3 + 2];
    }
    __syncthreads();

    const float cx = new_xyz[(size_t)gid * 3 + 0];
    const float cy = new_xyz[(size_t)gid * 3 + 1];
    const float cz = new_xyz[(size_t)gid * 3 + 2];

    int cnt = 0;
    for (int base = 0; base < N_; base += 32) {
        const int p = base + lane;
        const float dx = sx[p] - cx, dy = sy[p] - cy, dz = sz[p] - cz;
        const bool in = (dx * dx + dy * dy + dz * dz) < R2_;
#if __has_builtin(__builtin_amdgcn_ballot_w32)
        const unsigned m = __builtin_amdgcn_ballot_w32(in);
#else
        const unsigned m = (unsigned)__ballot(in);
#endif
        const int pos = cnt + __popc(m & ((1u << lane) - 1u));
        if (in && pos < NS_) s_hits[wv][pos] = p;
        cnt += __popc(m);                      // wave-uniform
        if (cnt >= NS_) break;                 // early exit (wave-uniform)
    }
    __syncthreads();                           // all threads reach this

    const int filled = cnt < NS_ ? cnt : NS_;
    const int first  = (filled > 0) ? s_hits[wv][0] : 0;
    const int v      = (lane < filled) ? s_hits[wv][lane] : first;
    idx_ws[(size_t)gid * NS_ + lane]  = v;
    out_idx[(size_t)gid * NS_ + lane] = v;
}

// ---------------------------------------------------------------------------
// 3) gather + bf16 convert, K padded 35 -> 64 with zeros
//    row = g*32 + j ; ch 0..2 = local xyz, 3..34 = point feats, 35..63 = 0
// ---------------------------------------------------------------------------
__global__ void gather_kernel(const float* __restrict__ xyz,
                              const float* __restrict__ points,
                              const float* __restrict__ new_xyz,
                              const int* __restrict__ idx_ws,
                              unsigned short* __restrict__ xbf) {
    const long tid = (long)blockIdx.x * blockDim.x + threadIdx.x; // rows*64
    const int  ch  = (int)(tid & 63);
    const long row = tid >> 6;
    const long g   = row >> 5;
    const int  j   = (int)(row & 31);
    const int  b   = (int)(g >> 10);
    const int  p   = idx_ws[g * NS_ + j];

    float v;
    if (ch < 3)
        v = xyz[((long)b * N_ + p) * 3 + ch] - new_xyz[g * 3 + ch];
    else if (ch < 3 + CPTS_)
        v = points[((long)b * N_ + p) * CPTS_ + (ch - 3)];
    else
        v = 0.0f;
    xbf[tid] = f2bf_u(v);
}

// ---------------------------------------------------------------------------
// 4) WMMA GEMM: Y[M, COUT] = Xbf[M, 64] @ W[CIN, COUT]
//    (conv bias folded out: training-mode BN subtracts the mean exactly).
//    Weights staged once per block in LDS as bf16 (K-padded). Each wave owns
//    one 16-wide N tile, holds both k-step B fragments in registers, and
//    sweeps MT M-tiles -> 2*MT back-to-back v_wmma_f32_16x16x32_bf16.
//    A fragment per ISA layout: lane<16 -> M=lane, k-base 0; lane>=16 -> +8;
//    VGPR 0..3 hold K=base..base+7, VGPR 4..7 hold K=base+16..base+23.
// ---------------------------------------------------------------------------
template <int CIN, int COUT, int MT>
__global__ void conv_wmma_kernel(const unsigned short* __restrict__ xbf,
                                 const float* __restrict__ W,
                                 float* __restrict__ Y) {
    constexpr int NT      = COUT / 16;      // N tiles (4 or 8)
    constexpr int MGROUPS = 8 / NT;         // waves stacked along M
    static_assert(NT * MGROUPS == 8, "block = 8 waves");

    __shared__ unsigned short lw[KPAD * COUT];
    for (int i = threadIdx.x; i < KPAD * COUT; i += 256) {
        const int k = i / COUT, n = i % COUT;       // shifts: COUT is pow2
        lw[i] = (k < CIN) ? f2bf_u(W[k * COUT + n]) : (unsigned short)0;
    }
    __syncthreads();

    const int wv   = threadIdx.x >> 5;
    const int lane = threadIdx.x & 31;
    const int ntile = wv % NT;                      // compile-time masks
    const int mg    = wv / NT;
    const int lm = lane & 15, hi = lane >> 4;
    const int n  = ntile * 16 + lm;

    // Build both k-step B fragments once; reuse across all M tiles.
    v16bf Bf[2];
#pragma unroll
    for (int ks = 0; ks < 2; ++ks) {
        const int kb = ks * 32 + hi * 8;
#pragma unroll
        for (int i = 0; i < 8; ++i) {
            const int k0 = kb + (i < 4 ? 2 * i : 16 + 2 * (i - 4));
            Bf[ks][2 * i]     = u2bf(lw[k0 * COUT + n]);
            Bf[ks][2 * i + 1] = u2bf(lw[(k0 + 1) * COUT + n]);
        }
    }

    const long mtile0 = ((long)blockIdx.x * MGROUPS + mg) * MT;
#pragma unroll
    for (int mt = 0; mt < MT; ++mt) {
        const long mtile = mtile0 + mt;
        const unsigned short* pa = xbf + (mtile * 16 + lm) * KPAD;
        v8f acc = {};
#pragma unroll
        for (int ks = 0; ks < 2; ++ks) {
            const int kb = ks * 32 + hi * 8;
            bfpack t;                                   // two 16B loads
            t.lo = *(const uint4v*)(pa + kb);           // K = kb .. kb+7
            t.hi = *(const uint4v*)(pa + kb + 16);      // K = kb+16 .. kb+23
            const v16bf A = __builtin_bit_cast(v16bf, t);
            acc = __builtin_amdgcn_wmma_f32_16x16x32_bf16(
                false, A, false, Bf[ks], (short)0, acc, false, false);
        }
        // C/D layout: lane N = lane&15; rows M = vgpr + (lane>=16 ? 8 : 0)
        const long mbase = mtile * 16 + hi * 8;
#pragma unroll
        for (int r = 0; r < 8; ++r)
            Y[(mbase + r) * COUT + n] = acc[r];
    }
}

// ---------------------------------------------------------------------------
// 5) BN stats, two deterministic stages, fully coalesced.
//    Stage 1: each block owns a row range; lane's channel = tid & (COUT-1),
//    so a wave reads 128B contiguous per row. Fixed-order LDS reduce.
// ---------------------------------------------------------------------------
template <int COUT>
__global__ void bnstats_partial_kernel(const float* __restrict__ Y,
                                       float* __restrict__ part,
                                       long rowsPerBlock) {
    constexpr int RSTEP = 256 / COUT;               // 4 or 2
    const int c  = threadIdx.x & (COUT - 1);
    const int rl = threadIdx.x / COUT;
    const long row0 = (long)blockIdx.x * rowsPerBlock;

    float s = 0.0f, q = 0.0f;
    for (long r = rl; r < rowsPerBlock; r += RSTEP) {
        const float v = Y[(row0 + r) * COUT + c];
        s += v; q += v * v;
    }
    __shared__ float ls[256], lq[256];
    ls[threadIdx.x] = s; lq[threadIdx.x] = q;
    __syncthreads();
    if (rl == 0) {
#pragma unroll
        for (int i = 1; i < RSTEP; ++i) {
            s += ls[c + i * COUT];
            q += lq[c + i * COUT];
        }
        part[(long)blockIdx.x * (2 * COUT) + c]        = s;
        part[(long)blockIdx.x * (2 * COUT) + COUT + c] = q;
    }
}

template <int COUT>
__global__ void bnstats_final_kernel(const float* __restrict__ part,
                                     float* __restrict__ stats,
                                     int nblocks, long M) {
    const int c = threadIdx.x;                      // blockDim == COUT
    float s = 0.0f, q = 0.0f;
    for (int p = 0; p < nblocks; ++p) {             // fixed order: deterministic
        s += part[(long)p * (2 * COUT) + c];
        q += part[(long)p * (2 * COUT) + COUT + c];
    }
    const float mean = s / (float)M;
    const float var  = q / (float)M - mean * mean;
    stats[c]       = mean;
    stats[128 + c] = rsqrtf(var + BN_EPS_);
}

// ---------------------------------------------------------------------------
// 6) BN + ReLU + bf16 convert (layers 1,2; Cout == 64 == next-layer K)
// ---------------------------------------------------------------------------
__global__ void bnrelu_kernel(const float* __restrict__ Y,
                              const float* __restrict__ stats,
                              const float* __restrict__ gamma,
                              const float* __restrict__ beta,
                              unsigned short* __restrict__ xbf) {
    const long tid = (long)blockIdx.x * blockDim.x + threadIdx.x;
    const int c = (int)(tid & 63);
    float v = (Y[tid] - stats[c]) * stats[128 + c] * gamma[c] + beta[c];
    v = fmaxf(v, 0.0f);
    xbf[tid] = f2bf_u(v);
}

// ---------------------------------------------------------------------------
// 7) Final layer: BN + ReLU + max-pool over ns=32 -> new_points
// ---------------------------------------------------------------------------
__global__ void bnmax_kernel(const float* __restrict__ Y,
                             const float* __restrict__ stats,
                             const float* __restrict__ gamma,
                             const float* __restrict__ beta,
                             float* __restrict__ out_pts) {
    const int g = blockIdx.x;        // group id, < 16384
    const int c = threadIdx.x;       // channel,  < 128
    const float m0 = stats[c], rs = stats[128 + c];
    const float ga = gamma[c], be = beta[c];
    float mx = 0.0f;                 // relu outputs are >= 0
#pragma unroll 4
    for (int j = 0; j < NS_; ++j) {
        float v = (Y[((long)g * NS_ + j) * 128 + c] - m0) * rs * ga + be;
        v = fmaxf(v, 0.0f);
        mx = fmaxf(mx, v);
    }
    out_pts[(long)g * 128 + c] = mx;
}

// ---------------------------------------------------------------------------
extern "C" void kernel_launch(void* const* d_in, const int* in_sizes, int n_in,
                              void* d_out, int out_size, void* d_ws, size_t ws_size,
                              hipStream_t stream) {
    const float* xyz    = (const float*)d_in[0];
    const float* points = (const float*)d_in[1];
    const float* w0 = (const float*)d_in[2];
    const float* g0 = (const float*)d_in[4];
    const float* e0 = (const float*)d_in[5];
    const float* w1 = (const float*)d_in[6];
    const float* g1 = (const float*)d_in[8];
    const float* e1 = (const float*)d_in[9];
    const float* w2 = (const float*)d_in[10];
    const float* g2 = (const float*)d_in[12];
    const float* e2 = (const float*)d_in[13];
    // biases b0/b1/b2 cancel exactly under training-mode batch norm.

    float* out_newxyz = (float*)d_out;                         // 16*1024*3
    float* out_newpts = out_newxyz + (size_t)NGROUP * 3;       // 16*1024*128
    int*   out_idx    = (int*)(out_newpts + (size_t)NGROUP * 128);

    char* wsb = (char*)d_ws;
    int*            idx_ws = (int*)wsb;                                   // 2 MiB
    unsigned short* xbf    = (unsigned short*)(wsb + (1ull << 21));       // 64 MiB
    float*          Ybuf   = (float*)(wsb + (1ull << 21) + (1ull << 26)); // 256 MiB
    char*           tail   = wsb + (1ull << 21) + (1ull << 26) + (1ull << 28);
    float*          stats  = (float*)tail;                                // 1 KiB (pad 4K)
    float*          part   = (float*)(tail + 4096);                       // 512 KiB

    const long M = MROWS;
    constexpr int SBLK = 512;                 // stats stage-1 blocks
    const long rowsPerBlk = M / SBLK;         // 1024

    // 1) FPS -> new_xyz
    fps_kernel<<<B_, 1024, 0, stream>>>(xyz, out_newxyz);
    // 2) ball query -> idx
    ball_kernel<<<NGROUP / 8, 256, 0, stream>>>(xyz, out_newxyz, idx_ws, out_idx);
    // 3) gather + bf16, K-pad to 64
    gather_kernel<<<(unsigned)((M * KPAD) / 256), 256, 0, stream>>>(
        xyz, points, out_newxyz, idx_ws, xbf);

    // layer 1: 35 -> 64   (tiles: 32768 M x 4 N; 8 tiles/block of 8 waves)
    conv_wmma_kernel<35, 64, 4><<<(unsigned)((M / 16) / 8), 256, 0, stream>>>(
        xbf, w0, Ybuf);
    bnstats_partial_kernel<64><<<SBLK, 256, 0, stream>>>(Ybuf, part, rowsPerBlk);
    bnstats_final_kernel<64><<<1, 64, 0, stream>>>(part, stats, SBLK, M);
    bnrelu_kernel<<<(unsigned)((M * 64) / 256), 256, 0, stream>>>(
        Ybuf, stats, g0, e0, xbf);

    // layer 2: 64 -> 64
    conv_wmma_kernel<64, 64, 4><<<(unsigned)((M / 16) / 8), 256, 0, stream>>>(
        xbf, w1, Ybuf);
    bnstats_partial_kernel<64><<<SBLK, 256, 0, stream>>>(Ybuf, part, rowsPerBlk);
    bnstats_final_kernel<64><<<1, 64, 0, stream>>>(part, stats, SBLK, M);
    bnrelu_kernel<<<(unsigned)((M * 64) / 256), 256, 0, stream>>>(
        Ybuf, stats, g1, e1, xbf);

    // layer 3: 64 -> 128 (each block: 1 m-group x 4 m-tiles)
    conv_wmma_kernel<64, 128, 4><<<(unsigned)((M / 16) / 4), 256, 0, stream>>>(
        xbf, w2, Ybuf);
    bnstats_partial_kernel<128><<<SBLK, 256, 0, stream>>>(Ybuf, part, rowsPerBlk);
    bnstats_final_kernel<128><<<1, 128, 0, stream>>>(part, stats, SBLK, M);
    bnmax_kernel<<<NGROUP, 128, 0, stream>>>(Ybuf, stats, g2, e2, out_newpts);
}
